// Decoder_10677288698614
// MI455X (gfx1250) — compile-verified, matrix operand order checked by reading, not run on hardware
//
#include <hip/hip_runtime.h>
#include <hip/hip_bf16.h>
#include <math.h>

typedef __attribute__((ext_vector_type(16))) __bf16 bf16x16;
typedef __attribute__((ext_vector_type(8)))  float  f32x8;

#define Bn    64
#define Tn    400
#define Hn    512
#define En    256
#define Vn    50000
#define NOOVn 100
#define TPn   50
#define H2n   1024
#define VEXTn 50100

// ---- d_out layout (flattened tuple, reference return order) ----
#define OFF_FINAL 0
#define OFF_H     3206400
#define OFF_C     3239168
#define OFF_CTE   3271936
#define OFF_AU    3337472
#define OFF_ET2   3363072
#define OFF_SUM   3428608
#define OFF_PREVS 3454208

// ---- workspace layout (float offsets) ----
#define WS_X       0
#define WS_STHAT   16384
#define WS_ENCB    81920
#define WS_DECB    147456
#define WS_ET      180224   // [B*T]  zeroed
#define WS_AUS     205824   // [B*T]  zeroed
#define WS_ET2     231424   // [B,H2] zeroed
#define WS_AT      296960
#define WS_CTD     322560
#define WS_PGEN    355328
#define WS_OUTV    355392
#define WS_RMAX    388160
#define WS_RSUM    388224
#define WS_LOGITS  388288   // [B,V]
#define WS_ZERO_N  116736   // et+aus+et2 contiguous

__device__ __forceinline__ float sigmoidf(float x) { return 1.0f / (1.0f + expf(-x)); }

// ---------------- zero accumulators ----------------
__global__ void k_zero(float* __restrict__ p, int n) {
  int i = blockIdx.x * blockDim.x + threadIdx.x;
  if (i < n) p[i] = 0.0f;
}

// ---------------- x = [x_t | ct_e] @ W_xc + b_xc ----------------
__global__ __launch_bounds__(256) void k_xctx(const float* __restrict__ xt, const float* __restrict__ cte,
                                              const float* __restrict__ Wxc, const float* __restrict__ bxc,
                                              float* __restrict__ xout) {
  __shared__ float row[En + H2n];
  int b = blockIdx.x, tid = threadIdx.x;
  for (int i = tid; i < En;  i += 256) row[i]      = xt[b * En + i];
  for (int i = tid; i < H2n; i += 256) row[En + i] = cte[b * H2n + i];
  __syncthreads();
  float acc = bxc[tid];
  for (int k = 0; k < En + H2n; ++k) acc += row[k] * Wxc[k * En + tid];
  xout[b * En + tid] = acc;
}

// ---------------- LSTM cell step ----------------
__global__ __launch_bounds__(512) void k_lstm(const float* __restrict__ x, const float* __restrict__ sh,
                                              const float* __restrict__ sc,
                                              const float* __restrict__ Wih, const float* __restrict__ Whh,
                                              const float* __restrict__ bih, const float* __restrict__ bhh,
                                              float* __restrict__ out, float* __restrict__ sthat) {
  __shared__ float lx[En];
  __shared__ float lh[Hn];
  int b = blockIdx.x, j = threadIdx.x;
  if (j < En) lx[j] = x[b * En + j];
  lh[j] = sh[b * Hn + j];
  __syncthreads();
  float gi = bih[j] + bhh[j];
  float gf = bih[j + Hn] + bhh[j + Hn];
  float gg = bih[j + 2 * Hn] + bhh[j + 2 * Hn];
  float go = bih[j + 3 * Hn] + bhh[j + 3 * Hn];
  for (int k = 0; k < En; ++k) {
    float v = lx[k]; const float* w = &Wih[k * 4 * Hn + j];
    gi += v * w[0]; gf += v * w[Hn]; gg += v * w[2 * Hn]; go += v * w[3 * Hn];
  }
  for (int k = 0; k < Hn; ++k) {
    float v = lh[k]; const float* w = &Whh[k * 4 * Hn + j];
    gi += v * w[0]; gf += v * w[Hn]; gg += v * w[2 * Hn]; go += v * w[3 * Hn];
  }
  float c = sigmoidf(gf) * sc[b * Hn + j] + sigmoidf(gi) * tanhf(gg);
  float h = sigmoidf(go) * tanhf(c);
  out[OFF_H + b * Hn + j] = h;
  out[OFF_C + b * Hn + j] = c;
  sthat[b * H2n + j]      = h;
  sthat[b * H2n + Hn + j] = c;
}

// ---------------- encBias = st_hat@We_s+be_s ; decBias = h@Wd_s+bd_s ----------------
__global__ __launch_bounds__(1024) void k_bias(const float* __restrict__ sthat,
                                               const float* __restrict__ Wes, const float* __restrict__ bes,
                                               const float* __restrict__ Wds, const float* __restrict__ bds,
                                               float* __restrict__ encB, float* __restrict__ decB) {
  __shared__ float ls[H2n];
  int b = blockIdx.x, j = threadIdx.x;
  ls[j] = sthat[b * H2n + j];
  __syncthreads();
  float acc = bes[j];
  for (int k = 0; k < H2n; ++k) acc += ls[k] * Wes[k * H2n + j];
  encB[b * H2n + j] = acc;
  if (j < Hn) {
    float a2 = bds[j];
    for (int k = 0; k < Hn; ++k) a2 += ls[k] * Wds[k * Hn + j];  // h = ls[0:512]
    decB[b * Hn + j] = a2;
  }
}

// ---------------- fused dual WMMA GEMM over enc_out + tanh epilogues ----------------
// rows = B*T (25600), cols = H2 (1024), K = H2. Block: 8 waves, 16 rows x 128 cols.
// A tiles (16x32 f32, 2KB) are double-buffered in LDS via GLOBAL_LOAD_ASYNC_TO_LDS_B128,
// issued by wave 0 (4 per-lane b128 ops per tile), tracked with ASYNCcnt.
__global__ __launch_bounds__(256) void k_dual_gemm(const float* __restrict__ enc,
                                                   const float* __restrict__ Weh, const float* __restrict__ Wsh,
                                                   const float* __restrict__ encB, const float* __restrict__ bsh,
                                                   const float* __restrict__ ve,  const float* __restrict__ vs1,
                                                   const float* __restrict__ vs2,
                                                   float* __restrict__ et_acc, float* __restrict__ aus_acc,
                                                   float* __restrict__ et2_acc) {
  __shared__ __align__(16) float lAf[2][16][32];  // double-buffered f32 A tile
  const int tid  = threadIdx.x;
  const int wave = tid >> 5, lane = tid & 31;
  const int rowbase = blockIdx.x * 16;
  const int colbase = (blockIdx.y * 8 + wave) * 16;
  const int b  = rowbase / Tn;
  const int t0 = rowbase - b * Tn;            // tiles never straddle b (400 % 16 == 0)
  const int lcol  = lane & 15;
  const int khalf = (lane >> 4) * 8;
  const unsigned ldsBase = (unsigned)(size_t)&lAf[0][0][0];
  f32x8 accE = {}; f32x8 accS = {};

  if (wave == 0) {                            // prologue: DMA tile 0 into buffer 0
#pragma unroll
    for (int inst = 0; inst < 4; ++inst) {
      unsigned o = (unsigned)inst * 512u + (unsigned)lane * 16u;      // byte offset in 2KB tile
      const float* g = enc + (size_t)(rowbase + (int)(o >> 7)) * H2n + (int)((o & 127u) >> 2);
      unsigned l = ldsBase + o;
      asm volatile("global_load_async_to_lds_b128 %0, %1, off" :: "v"(l), "v"(g) : "memory");
    }
  }

  for (int step = 0; step < H2n / 32; ++step) {
    const int kb  = step * 32;
    const int buf = step & 1;
    if (wave == 0) {
      if (step + 1 < H2n / 32) {              // DMA next tile into other buffer, then wait for oldest
#pragma unroll
        for (int inst = 0; inst < 4; ++inst) {
          unsigned o = (unsigned)inst * 512u + (unsigned)lane * 16u;
          const float* g = enc + (size_t)(rowbase + (int)(o >> 7)) * H2n + (kb + 32) + (int)((o & 127u) >> 2);
          unsigned l = ldsBase + (unsigned)(buf ^ 1) * 2048u + o;
          asm volatile("global_load_async_to_lds_b128 %0, %1, off" :: "v"(l), "v"(g) : "memory");
        }
        asm volatile("s_wait_asynccnt 0x4" ::: "memory");  // tile `step` complete, next may be in flight
      } else {
        asm volatile("s_wait_asynccnt 0x0" ::: "memory");  // last tile: full drain
      }
    }
    __syncthreads();                          // LDS tile visible to all 8 waves
    bf16x16 a, bE, bS;
#pragma unroll
    for (int e = 0; e < 16; ++e) {
      int k = khalf + (e & 7) + ((e >> 3) << 4);   // ISA 16-bit A/B K pattern
      a[e]  = (__bf16)lAf[buf][lcol][k];
      bE[e] = (__bf16)Weh[(size_t)(kb + k) * H2n + colbase + lcol];
      bS[e] = (__bf16)Wsh[(size_t)(kb + k) * H2n + colbase + lcol];
    }
    accE = __builtin_amdgcn_wmma_f32_16x16x32_bf16(false, a, false, bE, (short)0, accE, false, false);
    accS = __builtin_amdgcn_wmma_f32_16x16x32_bf16(false, a, false, bS, (short)0, accS, false, false);
    __syncthreads();                          // all reads done before this buffer is DMA-overwritten
  }

  // epilogue: et = sum_n tanh(.)*ve ; au_raw = sum_n tanh(.)*vs1 ; et2_raw[b,n] += sum_t tanh(.)*vs2[t]
  const int col   = colbase + lcol;
  const float veC = ve[col], vs1C = vs1[col], bsC = bsh[col];
  const float ebC = encB[(size_t)b * H2n + col];
  float e2local = 0.0f;
#pragma unroll
  for (int r = 0; r < 8; ++r) {
    int M   = r + ((lane >> 4) << 3);
    int row = rowbase + M;
    float tE = tanhf(accE[r] + ebC);
    float tS = tanhf(accS[r] + bsC);
    float cE = tE * veC;
    float cA = tS * vs1C;
#pragma unroll
    for (int off = 8; off >= 1; off >>= 1) {     // 16-lane (N-dim) butterfly
      cE += __shfl_xor(cE, off, 16);
      cA += __shfl_xor(cA, off, 16);
    }
    if (lcol == 0) {
      atomicAdd(&et_acc[row],  cE);
      atomicAdd(&aus_acc[row], cA);
    }
    e2local += tS * vs2[t0 + M];
  }
  atomicAdd(&et2_acc[(size_t)b * H2n + col], e2local);
}

// ---------------- encoder attention normalization + sum_new ----------------
__global__ __launch_bounds__(512) void k_encattn(const float* __restrict__ et_acc, const float* __restrict__ sumt,
                                                 const float* __restrict__ mask, float* __restrict__ out,
                                                 float* __restrict__ at_ws) {
  __shared__ float red[512];
  int b = blockIdx.x, t = threadIdx.x;
  float atun = 0.0f;
  if (t < Tn) {
    float ex = expf(et_acc[b * Tn + t]);
    float st = sumt[b * Tn + t];
    out[OFF_SUM + b * Tn + t] = st + ex;
    atun = (ex / st) * mask[b * Tn + t];
  }
  red[t] = atun;
  __syncthreads();
  for (int s = 256; s > 0; s >>= 1) { if (t < s) red[t] += red[t + s]; __syncthreads(); }
  float tot = red[0];
  if (t < Tn) at_ws[b * Tn + t] = atun / tot;
}

// ---------------- ct_e_new = at @ enc_out ----------------
__global__ __launch_bounds__(256) void k_cte(const float* __restrict__ at_ws, const float* __restrict__ enc,
                                             float* __restrict__ out) {
  int b = blockIdx.x, tid = threadIdx.x;
  float a0 = 0, a1 = 0, a2 = 0, a3 = 0;
  for (int t = 0; t < Tn; ++t) {
    float a = at_ws[b * Tn + t];
    const float* e = &enc[((size_t)b * Tn + t) * H2n];
    a0 += a * e[tid]; a1 += a * e[tid + 256]; a2 += a * e[tid + 512]; a3 += a * e[tid + 768];
  }
  float* o = &out[OFF_CTE + b * H2n];
  o[tid] = a0; o[tid + 256] = a1; o[tid + 512] = a2; o[tid + 768] = a3;
}

// ---------------- au (sigmoid+mask+normalize) ; et2 sigmoid ----------------
__global__ __launch_bounds__(1024) void k_sentfin(const float* __restrict__ aus, const float* __restrict__ et2acc,
                                                  const float* __restrict__ mask, float* __restrict__ out) {
  __shared__ float red[1024];
  int b = blockIdx.x, j = threadIdx.x;
  float v = 0.0f;
  if (j < Tn) v = sigmoidf(aus[b * Tn + j]) * mask[b * Tn + j];
  red[j] = v;
  __syncthreads();
  for (int s = 512; s > 0; s >>= 1) { if (j < s) red[j] += red[j + s]; __syncthreads(); }
  float tot = red[0];
  if (j < Tn) out[OFF_AU + b * Tn + j] = v / tot;
  out[OFF_ET2 + b * H2n + j] = sigmoidf(et2acc[b * H2n + j]);
}

// ---------------- intra-decoder attention ----------------
__global__ __launch_bounds__(512) void k_decattn(const float* __restrict__ prevs, const float* __restrict__ Wdp,
                                                 const float* __restrict__ decB, const float* __restrict__ vd,
                                                 float* __restrict__ ctd) {
  __shared__ float red[512];
  __shared__ float evals[TPn];
  __shared__ float lh[Hn];
  int b = blockIdx.x, j = threadIdx.x;
  float db = decB[b * Hn + j];
  for (int tp = 0; tp < TPn; ++tp) {
    lh[j] = prevs[((size_t)b * TPn + tp) * Hn + j];
    __syncthreads();
    float acc = db;
    for (int k = 0; k < Hn; ++k) acc += lh[k] * Wdp[k * Hn + j];
    red[j] = tanhf(acc) * vd[j];
    __syncthreads();
    for (int s = 256; s > 0; s >>= 1) { if (j < s) red[j] += red[j + s]; __syncthreads(); }
    if (j == 0) evals[tp] = red[0];
    __syncthreads();
  }
  if (j == 0) {
    float mx = -1e30f;
    for (int tp = 0; tp < TPn; ++tp) mx = fmaxf(mx, evals[tp]);
    float s = 0.0f;
    for (int tp = 0; tp < TPn; ++tp) { evals[tp] = expf(evals[tp] - mx); s += evals[tp]; }
    for (int tp = 0; tp < TPn; ++tp) evals[tp] /= s;
  }
  __syncthreads();
  float acc = 0.0f;
  for (int tp = 0; tp < TPn; ++tp) acc += evals[tp] * prevs[((size_t)b * TPn + tp) * Hn + j];
  ctd[b * Hn + j] = acc;
}

// ---------------- p_gen + out_vec = concat(h,ct_e,ct_d,et2)@W_V + b_V ----------------
__global__ __launch_bounds__(512) void k_pgen_outv(const float* __restrict__ out, const float* __restrict__ ctd,
                                                   const float* __restrict__ sthat,
                                                   const float* __restrict__ wpgcte, const float* __restrict__ wpgctd,
                                                   const float* __restrict__ wpgst,  const float* __restrict__ wpgem,
                                                   const float* __restrict__ bpgst,
                                                   const float* __restrict__ WV, const float* __restrict__ bV,
                                                   float* __restrict__ pgen, float* __restrict__ outvec) {
  __shared__ float vec[6 * Hn];
  __shared__ float red[512];
  int b = blockIdx.x, j = threadIdx.x;
  for (int i = j; i < 6 * Hn; i += 512) {
    float v;
    if (i < Hn)                 v = out[OFF_H   + b * Hn  + i];
    else if (i < Hn + H2n)      v = out[OFF_CTE + b * H2n + (i - Hn)];
    else if (i < 2 * Hn + H2n)  v = ctd[b * Hn + (i - Hn - H2n)];
    else                        v = out[OFF_ET2 + b * H2n + (i - 2 * Hn - H2n)];
    vec[i] = v;
  }
  __syncthreads();
  float p = 0.0f;
  for (int k = j; k < H2n; k += 512)
    p += vec[Hn + k] * wpgcte[k] + vec[2 * Hn + H2n + k] * wpgem[k] + sthat[b * H2n + k] * wpgst[k];
  p += vec[Hn + H2n + j] * wpgctd[j];
  red[j] = p;
  __syncthreads();
  for (int s = 256; s > 0; s >>= 1) { if (j < s) red[j] += red[j + s]; __syncthreads(); }
  if (j == 0) pgen[b] = sigmoidf(red[0] + bpgst[0]);
  float acc = bV[j];
  for (int k = 0; k < 6 * Hn; ++k) acc += vec[k] * WV[k * Hn + j];
  outvec[b * Hn + j] = acc;
}

// ---------------- vocab WMMA GEMM: logits = out_vec @ W_V1 + b_V1 ----------------
// Single pass over W_V1: each wave keeps 4 row-tile accumulators (M=64) and reuses
// every B fragment 4x, so the 102MB weight stream is read exactly once.
__global__ __launch_bounds__(256) void k_vocab_gemm(const float* __restrict__ outvec, const float* __restrict__ Wv1,
                                                    const float* __restrict__ bv1, float* __restrict__ logits) {
  const int tid = threadIdx.x, wave = tid >> 5, lane = tid & 31;
  const int colTile = blockIdx.x * 8 + wave;
  if (colTile >= Vn / 16) return;             // wave-uniform: EXEC stays all-ones for WMMA
  const int colbase = colTile * 16;
  const int lcol = lane & 15, khalf = (lane >> 4) * 8;
  f32x8 acc[4] = {{}, {}, {}, {}};
  for (int kb = 0; kb < Hn; kb += 32) {
    bf16x16 bb;
#pragma unroll
    for (int e = 0; e < 16; ++e) {
      int k = khalf + (e & 7) + ((e >> 3) << 4);
      bb[e] = (__bf16)Wv1[(size_t)(kb + k) * Vn + colbase + lcol];
    }
    if (kb + 32 < Hn) __builtin_prefetch(&Wv1[(size_t)(kb + 32) * Vn + colbase + lcol], 0, 1);
#pragma unroll
    for (int ry = 0; ry < 4; ++ry) {
      bf16x16 a;
#pragma unroll
      for (int e = 0; e < 16; ++e) {
        int k = khalf + (e & 7) + ((e >> 3) << 4);
        a[e] = (__bf16)outvec[(ry * 16 + lcol) * Hn + kb + k];
      }
      acc[ry] = __builtin_amdgcn_wmma_f32_16x16x32_bf16(false, a, false, bb, (short)0, acc[ry], false, false);
    }
  }
  const int col = colbase + lcol;
  const float bias = bv1[col];
#pragma unroll
  for (int ry = 0; ry < 4; ++ry) {
#pragma unroll
    for (int r = 0; r < 8; ++r) {
      int M = ry * 16 + r + ((lane >> 4) << 3);
      logits[(size_t)M * Vn + col] = acc[ry][r] + bias;
    }
  }
}

// ---------------- per-row softmax statistics ----------------
__global__ __launch_bounds__(1024) void k_smstats(const float* __restrict__ logits, float* __restrict__ rmax,
                                                  float* __restrict__ rsum) {
  __shared__ float red[1024];
  int b = blockIdx.x, j = threadIdx.x;
  float mx = -1e30f;
  for (int v = j; v < Vn; v += 1024) mx = fmaxf(mx, logits[(size_t)b * Vn + v]);
  red[j] = mx; __syncthreads();
  for (int s = 512; s > 0; s >>= 1) { if (j < s) red[j] = fmaxf(red[j], red[j + s]); __syncthreads(); }
  float m = red[0];
  __syncthreads();
  float sm = 0.0f;
  for (int v = j; v < Vn; v += 1024) sm += expf(logits[(size_t)b * Vn + v] - m);
  red[j] = sm; __syncthreads();
  for (int s = 512; s > 0; s >>= 1) { if (j < s) red[j] += red[j + s]; __syncthreads(); }
  if (j == 0) { rmax[b] = m; rsum[b] = red[0]; }
}

// ---------------- final dist = p_gen*softmax ; OOV zeros ----------------
__global__ void k_final(const float* __restrict__ logits, const float* __restrict__ pgen,
                        const float* __restrict__ rmax, const float* __restrict__ rsum,
                        float* __restrict__ out) {
  int i = blockIdx.x * blockDim.x + threadIdx.x;
  if (i >= Bn * VEXTn) return;
  int b = i / VEXTn, v = i - b * VEXTn;
  float val = 0.0f;
  if (v < Vn) val = pgen[b] * expf(logits[(size_t)b * Vn + v] - rmax[b]) / rsum[b];
  out[OFF_FINAL + i] = val;
}

// ---------------- pointer scatter-add ----------------
__global__ __launch_bounds__(512) void k_scatter(const float* __restrict__ at_ws, const float* __restrict__ pgen,
                                                 const int* __restrict__ ebev, float* __restrict__ out) {
  int b = blockIdx.x, t = threadIdx.x;
  if (t < Tn) {
    int idx = ebev[b * Tn + t];
    float add = (1.0f - pgen[b]) * at_ws[b * Tn + t];
    atomicAdd(&out[OFF_FINAL + (size_t)b * VEXTn + idx], add);
  }
}

// ---------------- prev_s_new = concat(prev_s, h) ----------------
__global__ void k_prevs(const float* __restrict__ prevs, float* __restrict__ out) {
  int i = blockIdx.x * blockDim.x + threadIdx.x;
  const int TOT = Bn * (TPn + 1) * Hn;
  if (i >= TOT) return;
  int b  = i / ((TPn + 1) * Hn);
  int r  = i - b * (TPn + 1) * Hn;
  int tp = r / Hn, j = r - tp * Hn;
  float v = (tp < TPn) ? prevs[((size_t)b * TPn + tp) * Hn + j] : out[OFF_H + b * Hn + j];
  out[OFF_PREVS + i] = v;
}

extern "C" void kernel_launch(void* const* d_in, const int* in_sizes, int n_in,
                              void* d_out, int out_size, void* d_ws, size_t ws_size,
                              hipStream_t stream) {
  const float* x_t   = (const float*)d_in[0];
  const float* s_h   = (const float*)d_in[1];
  const float* s_c   = (const float*)d_in[2];
  const float* enc   = (const float*)d_in[3];
  const float* mask  = (const float*)d_in[4];
  const float* ct_e  = (const float*)d_in[5];
  const float* sumt  = (const float*)d_in[7];
  const float* prevs = (const float*)d_in[8];
  const int*   ebev  = (const int*)d_in[9];
  const float* Wxc = (const float*)d_in[10];
  const float* bxc = (const float*)d_in[11];
  const float* Wih = (const float*)d_in[12];
  const float* Whh = (const float*)d_in[13];
  const float* bih = (const float*)d_in[14];
  const float* bhh = (const float*)d_in[15];
  const float* Weh = (const float*)d_in[16];
  const float* Wes = (const float*)d_in[17];
  const float* bes = (const float*)d_in[18];
  const float* ve  = (const float*)d_in[19];
  const float* Wsh = (const float*)d_in[20];
  const float* bsh = (const float*)d_in[21];
  const float* vs1 = (const float*)d_in[22];
  const float* vs2 = (const float*)d_in[23];
  const float* Wdp = (const float*)d_in[24];
  const float* Wds = (const float*)d_in[25];
  const float* bds = (const float*)d_in[26];
  const float* vd  = (const float*)d_in[27];
  const float* wpgem  = (const float*)d_in[28];
  const float* wpgcte = (const float*)d_in[29];
  const float* wpgctd = (const float*)d_in[30];
  const float* wpgst  = (const float*)d_in[31];
  const float* bpgst  = (const float*)d_in[32];
  const float* WV  = (const float*)d_in[33];
  const float* bV  = (const float*)d_in[34];
  const float* WV1 = (const float*)d_in[35];
  const float* bV1 = (const float*)d_in[36];
  float* out = (float*)d_out;
  float* ws  = (float*)d_ws;

  k_zero<<<(WS_ZERO_N + 255) / 256, 256, 0, stream>>>(ws + WS_ET, WS_ZERO_N);
  k_xctx<<<Bn, 256, 0, stream>>>(x_t, ct_e, Wxc, bxc, ws + WS_X);
  k_lstm<<<Bn, 512, 0, stream>>>(ws + WS_X, s_h, s_c, Wih, Whh, bih, bhh, out, ws + WS_STHAT);
  k_bias<<<Bn, 1024, 0, stream>>>(ws + WS_STHAT, Wes, bes, Wds, bds, ws + WS_ENCB, ws + WS_DECB);
  k_dual_gemm<<<dim3(Bn * Tn / 16, 8), 256, 0, stream>>>(enc, Weh, Wsh, ws + WS_ENCB, bsh, ve, vs1, vs2,
                                                         ws + WS_ET, ws + WS_AUS, ws + WS_ET2);
  k_encattn<<<Bn, 512, 0, stream>>>(ws + WS_ET, sumt, mask, out, ws + WS_AT);
  k_cte<<<Bn, 256, 0, stream>>>(ws + WS_AT, enc, out);
  k_sentfin<<<Bn, 1024, 0, stream>>>(ws + WS_AUS, ws + WS_ET2, mask, out);
  k_decattn<<<Bn, 512, 0, stream>>>(prevs, Wdp, ws + WS_DECB, vd, ws + WS_CTD);
  k_pgen_outv<<<Bn, 512, 0, stream>>>(out, ws + WS_CTD, ws + WS_STHAT, wpgcte, wpgctd, wpgst, wpgem, bpgst,
                                      WV, bV, ws + WS_PGEN, ws + WS_OUTV);
  k_vocab_gemm<<<dim3((Vn / 16 + 7) / 8), 256, 0, stream>>>(ws + WS_OUTV, WV1, bV1, ws + WS_LOGITS);
  k_smstats<<<Bn, 1024, 0, stream>>>(ws + WS_LOGITS, ws + WS_RMAX, ws + WS_RSUM);
  k_final<<<(Bn * VEXTn + 255) / 256, 256, 0, stream>>>(ws + WS_LOGITS, ws + WS_PGEN, ws + WS_RMAX,
                                                        ws + WS_RSUM, out);
  k_scatter<<<Bn, 512, 0, stream>>>(ws + WS_AT, ws + WS_PGEN, ebev, out);
  k_prevs<<<(Bn * (TPn + 1) * Hn + 255) / 256, 256, 0, stream>>>(prevs, out);
}
